// LSAAttention_34230889349531
// MI455X (gfx1250) — matrix operand order repaired and checked by reading, not compiled, not run on hardware
//
#include <hip/hip_runtime.h>

typedef __attribute__((ext_vector_type(16))) _Float16 v16h;
typedef __attribute__((ext_vector_type(8)))  float    v8f;

#define SEQ    2048
#define DIMH   64
#define NHEADS (4 * 12)
#define QTILE  128            // q rows per block (8 waves x 16)
#define WQT    16             // q rows per wave
#define CHUNK  64             // keys staged per iteration
#define NCHUNK (SEQ / CHUNK)
#define KP 72                 // LDS row stride (halves), padded vs 64 to dodge bank conflicts
#define VP 72
#define PP 72

__global__ __launch_bounds__(256) void lsa_attn_kernel(
    const float* __restrict__ Qg, const float* __restrict__ Kg,
    const float* __restrict__ Vg, const float* __restrict__ Tg,
    float* __restrict__ Og)
{
  __shared__ _Float16 sK[CHUNK * KP];      // K chunk, [key][d], f16
  __shared__ _Float16 sV[DIMH * VP];       // V chunk transposed, [d][key], f16
  __shared__ _Float16 sP[8 * WQT * PP];    // per-wave P round-trip, [row][key]

  const int qt   = blockIdx.x;
  const int head = blockIdx.y;
  const int tid  = threadIdx.x;
  const int wave = tid >> 5;
  const int lane = tid & 31;
  const int l16  = lane & 15;
  const int half = lane >> 4;
  const int h8   = half * 8;

  const size_t hbase = (size_t)head * SEQ * DIMH;
  const float* Qh = Qg + hbase;
  const float* Kh = Kg + hbase;
  const float* Vh = Vg + hbase;
  float*       Oh = Og + hbase;

  const float invt  = 1.0f / Tg[0];
  const int   qbase = qt * QTILE + wave * WQT;

  // ---- Q A-fragments: 16x64 f32 -> two 16x32 f16 frags, 1/T folded in.
  // ISA A-layout (16-bit, 16x32): lanes 0-15 row=lane, K={0..7,16..23}+pair-packed;
  // lanes 16-31 row=lane-16, K={8..15,24..31}.
  v16h a0, a1;
  {
    const float* qp = Qh + (size_t)(qbase + l16) * DIMH;
    #pragma unroll
    for (int i = 0; i < 8; ++i) {
      a0[i]     = (_Float16)(qp[h8 + i]      * invt);
      a0[8 + i] = (_Float16)(qp[16 + h8 + i] * invt);
      a1[i]     = (_Float16)(qp[32 + h8 + i] * invt);
      a1[8 + i] = (_Float16)(qp[48 + h8 + i] * invt);
    }
  }

  float m[8], lsum[8];
  v8f acc[4] = {};
  #pragma unroll
  for (int r = 0; r < 8; ++r) { m[r] = -INFINITY; lsum[r] = 0.0f; }

  _Float16* pwave = sP + wave * WQT * PP;

  for (int c = 0; c < NCHUNK; ++c) {
    __syncthreads();
    // ---- cooperative stage: K row-major f16, V transposed f16 (convert once/block)
    for (int i = tid; i < CHUNK * (DIMH / 4); i += 256) {
      const int key = i >> 4;
      const int d4  = (i & 15) << 2;
      const size_t g = (size_t)(c * CHUNK + key) * DIMH + d4;
      const float4 kv = *(const float4*)(Kh + g);
      _Float16* kd = sK + key * KP + d4;
      kd[0] = (_Float16)kv.x; kd[1] = (_Float16)kv.y;
      kd[2] = (_Float16)kv.z; kd[3] = (_Float16)kv.w;
      const float4 vv = *(const float4*)(Vh + g);
      sV[(d4 + 0) * VP + key] = (_Float16)vv.x;
      sV[(d4 + 1) * VP + key] = (_Float16)vv.y;
      sV[(d4 + 2) * VP + key] = (_Float16)vv.z;
      sV[(d4 + 3) * VP + key] = (_Float16)vv.w;
    }
    if (c + 1 < NCHUNK && tid < CHUNK) {   // warm L2 for next chunk (global_prefetch_b8)
      __builtin_prefetch(Kh + (size_t)((c + 1) * CHUNK + tid) * DIMH, 0, 1);
      __builtin_prefetch(Vh + (size_t)((c + 1) * CHUNK + tid) * DIMH, 0, 1);
    }
    __syncthreads();

    // ---- scores: S = (Q/T) x K^T, four 16x16 tiles, K-dim 64 = 2 WMMAs each.
    // B-layout (16-bit, 32x16): lane n<16 col=n rows K={0..7,16..23}; lanes 16-31 K={8..15,24..31}.
    v8f sc[4];
    #pragma unroll
    for (int j = 0; j < 4; ++j) {
      const _Float16* kp = sK + (j * 16 + l16) * KP;
      v16h b0, b1;
      #pragma unroll
      for (int i = 0; i < 8; ++i) { b0[i] = kp[h8 + i]; b0[8 + i] = kp[16 + h8 + i]; b1[i] = kp[32 + h8 + i]; b1[8 + i] = kp[48 + h8 + i]; }
      v8f s = {};
      s = __builtin_amdgcn_wmma_f32_16x16x32_f16(false, a0, false, b0, (short)0, s, false, false);
      s = __builtin_amdgcn_wmma_f32_16x16x32_f16(false, a1, false, b1, (short)0, s, false, false);
      sc[j] = s;
    }

    // ---- diagonal self-exclusion (C-layout: VGPR r -> row r+8*half, col = lane%16 + 16j)
    const int kb = c * CHUNK;
    if (qbase < kb + CHUNK && kb < qbase + WQT) {
      #pragma unroll
      for (int j = 0; j < 4; ++j)
        #pragma unroll
        for (int r = 0; r < 8; ++r)
          if (qbase + r + h8 == kb + j * 16 + l16) sc[j][r] = -INFINITY;
    }

    // ---- online softmax: row stats via 16-lane butterflies (wave32)
    float rmax[8], cf[8], rsum[8];
    #pragma unroll
    for (int r = 0; r < 8; ++r)
      rmax[r] = fmaxf(fmaxf(sc[0][r], sc[1][r]), fmaxf(sc[2][r], sc[3][r]));
    #pragma unroll
    for (int off = 1; off < 16; off <<= 1)
      #pragma unroll
      for (int r = 0; r < 8; ++r)
        rmax[r] = fmaxf(rmax[r], __shfl_xor(rmax[r], off, 32));
    #pragma unroll
    for (int r = 0; r < 8; ++r) {
      const float mn = fmaxf(m[r], rmax[r]);
      cf[r] = __expf(m[r] - mn);
      m[r]  = mn;
      rsum[r] = 0.0f;
    }
    #pragma unroll
    for (int j = 0; j < 4; ++j)
      #pragma unroll
      for (int r = 0; r < 8; ++r) {
        const float p = __expf(sc[j][r] - m[r]);
        sc[j][r] = p;
        rsum[r] += p;
      }
    #pragma unroll
    for (int off = 1; off < 16; off <<= 1)
      #pragma unroll
      for (int r = 0; r < 8; ++r)
        rsum[r] += __shfl_xor(rsum[r], off, 32);
    #pragma unroll
    for (int r = 0; r < 8; ++r) {
      lsum[r] = lsum[r] * cf[r] + rsum[r];
      #pragma unroll
      for (int t = 0; t < 4; ++t) acc[t][r] *= cf[r];
    }

    // ---- P: C-layout -> LDS row-major f16 (wave-private region, DS-ordered)
    #pragma unroll
    for (int j = 0; j < 4; ++j)
      #pragma unroll
      for (int r = 0; r < 8; ++r)
        pwave[(r + h8) * PP + j * 16 + l16] = (_Float16)sc[j][r];
    asm volatile("s_wait_dscnt 0" ::: "memory");

    // ---- reload P as A-fragments; O += P x V (four d-tiles x 2 k-halves)
    const _Float16* pr = pwave + l16 * PP;
    v16h p0, p1;
    #pragma unroll
    for (int i = 0; i < 8; ++i) {
      p0[i]     = pr[h8 + i];
      p0[8 + i] = pr[16 + h8 + i];
      p1[i]     = pr[32 + h8 + i];
      p1[8 + i] = pr[48 + h8 + i];
    }
    #pragma unroll
    for (int t = 0; t < 4; ++t) {
      const _Float16* vp = sV + (t * 16 + l16) * VP;
      v16h b0, b1;
      #pragma unroll
      for (int i = 0; i < 8; ++i) { b0[i] = vp[h8 + i]; b0[8 + i] = vp[16 + h8 + i]; b1[i] = vp[32 + h8 + i]; b1[8 + i] = vp[48 + h8 + i]; }
      acc[t] = __builtin_amdgcn_wmma_f32_16x16x32_f16(false, p0, false, b0, (short)0, acc[t], false, false);
      acc[t] = __builtin_amdgcn_wmma_f32_16x16x32_f16(false, p1, false, b1, (short)0, acc[t], false, false);
    }
  }

  // ---- finalize: O = acc / l, write f32 straight from C-layout
  #pragma unroll
  for (int r = 0; r < 8; ++r) {
    const float rl = 1.0f / lsum[r];
    #pragma unroll
    for (int t = 0; t < 4; ++t)
      Oh[(size_t)(qbase + r + h8) * DIMH + t * 16 + l16] = acc[t][r] * rl;
  }
}

extern "C" void kernel_launch(void* const* d_in, const int* in_sizes, int n_in,
                              void* d_out, int out_size, void* d_ws, size_t ws_size,
                              hipStream_t stream) {
  (void)in_sizes; (void)n_in; (void)out_size; (void)d_ws; (void)ws_size;
  const float* Q = (const float*)d_in[0];
  const float* K = (const float*)d_in[1];
  const float* V = (const float*)d_in[2];
  const float* T = (const float*)d_in[3];
  float* O = (float*)d_out;
  dim3 grid(SEQ / QTILE, NHEADS);
  lsa_attn_kernel<<<grid, 256, 0, stream>>>(Q, K, V, T, O);
}
